// Linear_Attention_80367428043392
// MI455X (gfx1250) — matrix operand order repaired and checked by reading, not compiled, not run on hardware
//
#include <hip/hip_runtime.h>
#include <hip/hip_bf16.h>
#include <stdint.h>

typedef __attribute__((ext_vector_type(16))) _Float16 v16h;
typedef __attribute__((ext_vector_type(8)))  _Float16 v8h;
typedef __attribute__((ext_vector_type(8)))  float    v8f;
typedef __attribute__((ext_vector_type(4)))  int      v4i;

#define DIMM   512
#define SEQB   8192        // B*N = 4*2048
#define BM     128
#define BN     256
#define BK     32
#define LDSS   40          // BK + 8 pad (halves); row stride 80B keeps 16B alignment
#define KVCH   8           // K-split chunks for the kv reduction

__device__ inline v16h mkfrag(v8h lo, v8h hi) {
  v16h r;
#pragma unroll
  for (int i = 0; i < 8; ++i) { r[i] = lo[i]; r[i + 8] = hi[i]; }
  return r;
}

__device__ inline v8f wmma_f16(v16h a, v16h b, v8f c) {
  return __builtin_amdgcn_wmma_f32_16x16x32_f16(false, a, false, b, (short)0, c, false, false);
}

// ---- async 16-byte global -> LDS copy (per-lane), tracked by ASYNCcnt ----
__device__ __forceinline__ void async_cp16(const _Float16* g, _Float16* l) {
#if __has_builtin(__builtin_amdgcn_global_load_async_to_lds_b128)
  __builtin_amdgcn_global_load_async_to_lds_b128(
      (__attribute__((address_space(1))) v4i*)(uintptr_t)g,
      (__attribute__((address_space(3))) v4i*)(uint32_t)(uintptr_t)l,
      0, 0);
#else
  uint32_t lo = (uint32_t)(uintptr_t)l;   // LDS byte address = low 32 bits of flat addr
  asm volatile("global_load_async_to_lds_b128 %0, %1, off"
               :: "v"(lo), "v"(g) : "memory");
#endif
}

__device__ __forceinline__ void wait_async0() {
#if __has_builtin(__builtin_amdgcn_s_wait_asynccnt)
  __builtin_amdgcn_s_wait_asynccnt(0);
#else
  asm volatile("s_wait_asynccnt 0x0" ::: "memory");
#endif
}

// ---------------------------------------------------------------- generic WMMA GEMM, C = A(f16,MxK) * Bt(f16,NxK)^T
struct GemmP {
  const _Float16* A;  const _Float16* Bt;
  int N, K, lda, ldbt;
  const float *b0, *b1, *b2;   // biases
  const float *resid;          // f32 residual (MODE 1)
  float *outF;                 // f32 output (MODE 1,3)
  _Float16 *o0, *o1, *o2;      // f16 outputs
};

// MODE 0: QKV (+elu epilogue, scatter k/v transposed)  MODE 1: Wo (+bias+residual -> f32)
// MODE 2: FFN1 (+bias, relu -> f16)                    MODE 3: FFN2 (+bias, += f32 out)
template <int MODE>
__global__ __launch_bounds__(256, 1)
void gemm_kernel(GemmP p) {
  __shared__ _Float16 As[2 * BM * LDSS];
  __shared__ _Float16 Bs[2 * BN * LDSS];
  const int tid = threadIdx.x;
  const int wave = tid >> 5, lane = tid & 31;
  const int wm = wave >> 2, wn = wave & 3;     // 2 x 4 wave grid
  const int lr = lane & 15, ks = lane >> 4;
  const int m0 = blockIdx.y * BM;
  const int n0 = blockIdx.x * BN;

  v8f acc[4][4] = {};

  // async-stage one K-tile (A: 2 chunks/thread, B: 4 chunks/thread; all b128)
  auto issue = [&](int buf, int k0) {
    _Float16* As_ = As + buf * (BM * LDSS);
    _Float16* Bs_ = Bs + buf * (BN * LDSS);
#pragma unroll
    for (int j = 0; j < 2; ++j) {
      int ch = tid + j * 256;
      int r = ch >> 2, c = (ch & 3) * 8;
      async_cp16(p.A + (size_t)(m0 + r) * p.lda + k0 + c, &As_[r * LDSS + c]);
    }
#pragma unroll
    for (int j = 0; j < 4; ++j) {
      int ch = tid + j * 256;
      int r = ch >> 2, c = (ch & 3) * 8;
      async_cp16(p.Bt + (size_t)(n0 + r) * p.ldbt + k0 + c, &Bs_[r * LDSS + c]);
    }
  };

  const int nk = p.K / BK;
  issue(0, 0);
  for (int i = 0; i < nk; ++i) {
    wait_async0();            // this wave's copies for stage i landed
    __syncthreads();          // all waves' copies landed; all waves done reading stage i-1
    if (i + 1 < nk) issue((i + 1) & 1, (i + 1) * BK);   // overlap with compute below

    const _Float16* As_ = As + (i & 1) * (BM * LDSS);
    const _Float16* Bs_ = Bs + (i & 1) * (BN * LDSS);
    v16h aF[4], bF[4];
#pragma unroll
    for (int mt = 0; mt < 4; ++mt) {
      const _Float16* s = &As_[(wm * 64 + mt * 16 + lr) * LDSS + ks * 8];
      aF[mt] = mkfrag(*(const v8h*)s, *(const v8h*)(s + 16));
    }
#pragma unroll
    for (int nt = 0; nt < 4; ++nt) {
      const _Float16* s = &Bs_[(wn * 64 + nt * 16 + lr) * LDSS + ks * 8];
      bF[nt] = mkfrag(*(const v8h*)s, *(const v8h*)(s + 16));
    }
#pragma unroll
    for (int mt = 0; mt < 4; ++mt)
#pragma unroll
      for (int nt = 0; nt < 4; ++nt)
        acc[mt][nt] = wmma_f16(aF[mt], bF[nt], acc[mt][nt]);
  }

#pragma unroll
  for (int mt = 0; mt < 4; ++mt) {
#pragma unroll
    for (int nt = 0; nt < 4; ++nt) {
      int n = n0 + wn * 64 + nt * 16 + lr;
      int mbase = m0 + wm * 64 + mt * 16 + ks * 8;
#pragma unroll
      for (int r = 0; r < 8; ++r) {
        int m = mbase + r;
        float a = acc[mt][nt][r];
        if constexpr (MODE == 0) {
          if (n < 512) {                               // Q: (x/8) -> elu+1
            float t = (a + p.b0[n]) * 0.125f;
            t = t > 0.f ? t + 1.f : __expf(t);
            p.o0[(size_t)m * 512 + n] = (_Float16)t;
          } else if (n < 1024) {                       // K: elu+1, write kT[(64*bh+d)*2048+nl]
            int c = n - 512;
            float t = a + p.b1[c];
            t = t > 0.f ? t + 1.f : __expf(t);
            p.o1[((size_t)(m >> 11) * 512 + c) * 2048 + (m & 2047)] = (_Float16)t;
          } else {                                     // V transposed
            int c = n - 1024;
            p.o2[((size_t)(m >> 11) * 512 + c) * 2048 + (m & 2047)] = (_Float16)(a + p.b2[c]);
          }
        } else if constexpr (MODE == 1) {
          p.outF[(size_t)m * 512 + n] = a + p.b0[n] + p.resid[(size_t)m * 512 + n];
        } else if constexpr (MODE == 2) {
          float t = a + p.b0[n];
          p.o0[(size_t)m * p.N + n] = (_Float16)(t > 0.f ? t : 0.f);
        } else {
          p.outF[(size_t)m * 512 + n] += a + p.b0[n];
        }
      }
    }
  }
}

// force early emission so the disasm snippet shows the async+wmma pipeline
template __global__ void gemm_kernel<0>(GemmP);
template __global__ void gemm_kernel<1>(GemmP);
template __global__ void gemm_kernel<2>(GemmP);
template __global__ void gemm_kernel<3>(GemmP);

// ---------------------------------------------------------------- weights: transpose + cast
// src is K x N (row-major f32), dst is N x K (row-major f16) : dst[n*K+k] = src[k*N+n]
__global__ void tcast_kernel(const float* __restrict__ src, _Float16* __restrict__ dst,
                             int K, int N) {
  int o = blockIdx.x * 256 + threadIdx.x;
  int n = o / K;
  int k = o - n * K;
  dst[o] = (_Float16)src[(size_t)k * N + n];
}

// ---------------------------------------------------------------- layernorm (512 cols), f32 in -> f16 out
__global__ __launch_bounds__(256, 1)
void ln_kernel(const float* __restrict__ x, const float* __restrict__ g,
               const float* __restrict__ be, _Float16* __restrict__ out) {
  int wave = threadIdx.x >> 5, lane = threadIdx.x & 31;
  int row = blockIdx.x * 8 + wave;
  const float* xp = x + (size_t)row * DIMM + lane * 16;
  float v[16];
  float s = 0.f, sq = 0.f;
#pragma unroll
  for (int j = 0; j < 16; ++j) { v[j] = xp[j]; s += v[j]; sq += v[j] * v[j]; }
#pragma unroll
  for (int off = 16; off > 0; off >>= 1) { s += __shfl_xor(s, off); sq += __shfl_xor(sq, off); }
  float mean = s * (1.f / DIMM);
  float var  = sq * (1.f / DIMM) - mean * mean;
  float rstd = rsqrtf(var + 1e-5f);
  _Float16* op = out + (size_t)row * DIMM + lane * 16;
  const float* gp = g + lane * 16;
  const float* bp = be + lane * 16;
#pragma unroll
  for (int j = 0; j < 16; ++j)
    op[j] = (_Float16)((v[j] - mean) * rstd * gp[j] + bp[j]);
}

// ---------------------------------------------------------------- kv partial: per (bh, chunk) 64x64 over 256 of K=2048
// also emits per-column partial sums of k (for the denominator)
__global__ __launch_bounds__(128, 1)
void kv_partial_kernel(const _Float16* __restrict__ kT, const _Float16* __restrict__ vT,
                       float* __restrict__ kvp, float* __restrict__ ksp) {
  int bh = blockIdx.x, chk = blockIdx.y;
  int wave = threadIdx.x >> 5, lane = threadIdx.x & 31;
  int lr = lane & 15, ks = lane >> 4;
  size_t cb = (size_t)bh * 64 * 2048;
  int d0 = wave * 16;
  int kbeg = chk * (2048 / KVCH);
  v8f acc[4] = {};
  for (int k0 = kbeg; k0 < kbeg + 2048 / KVCH; k0 += 32) {
    const _Float16* ap = kT + cb + (size_t)(d0 + lr) * 2048 + k0 + ks * 8;
    v16h aF = mkfrag(*(const v8h*)ap, *(const v8h*)(ap + 16));
    v16h bF[4];
#pragma unroll
    for (int et = 0; et < 4; ++et) {
      const _Float16* bp = vT + cb + (size_t)(et * 16 + lr) * 2048 + k0 + ks * 8;
      bF[et] = mkfrag(*(const v8h*)bp, *(const v8h*)(bp + 16));
    }
#pragma unroll
    for (int et = 0; et < 4; ++et)
      acc[et] = wmma_f16(aF, bF[et], acc[et]);
  }
  float* outp = kvp + ((size_t)chk * 32 + bh) * 4096;   // transposed: [e*64 + d]
#pragma unroll
  for (int et = 0; et < 4; ++et)
#pragma unroll
    for (int r = 0; r < 8; ++r)
      outp[(et * 16 + lr) * 64 + d0 + ks * 8 + r] = acc[et][r];

  if (threadIdx.x < 64) {                               // partial column-sum of k
    const _Float16* p = kT + cb + (size_t)threadIdx.x * 2048 + kbeg;
    float s = 0.f;
    for (int i = 0; i < 2048 / KVCH; i += 8) {
      v8h v = *(const v8h*)(p + i);
#pragma unroll
      for (int j = 0; j < 8; ++j) s += (float)v[j];
    }
    ksp[((size_t)chk * 32 + bh) * 64 + threadIdx.x] = s;
  }
}

// ---------------------------------------------------------------- deterministic fixed-order reduction of partials
__global__ __launch_bounds__(256, 1)
void kv_reduce_kernel(const float* __restrict__ kvp, const float* __restrict__ ksp,
                      _Float16* __restrict__ kvT, float* __restrict__ ksum) {
  int bh = blockIdx.x, tid = threadIdx.x;
#pragma unroll
  for (int t = 0; t < 16; ++t) {
    int j = t * 256 + tid;                              // 0..4095, coalesced
    float s = 0.f;
#pragma unroll
    for (int chk = 0; chk < KVCH; ++chk) s += kvp[((size_t)chk * 32 + bh) * 4096 + j];
    kvT[(size_t)bh * 4096 + j] = (_Float16)s;
  }
  if (tid < 64) {
    float s = 0.f;
#pragma unroll
    for (int chk = 0; chk < KVCH; ++chk) s += ksp[((size_t)chk * 32 + bh) * 64 + tid];
    ksum[bh * 64 + tid] = s;
  }
}

// ---------------------------------------------------------------- att = (q @ kv) / (q . ksum + 1e-5)
__global__ __launch_bounds__(256, 1)
void attn_kernel(const _Float16* __restrict__ q, const _Float16* __restrict__ kvT,
                 const float* __restrict__ ksum, _Float16* __restrict__ att) {
  __shared__ float sden[128];
  int h = blockIdx.x;
  int m0 = blockIdx.y * 128;
  int bh = (m0 >> 11) * 8 + h;
  int tid = threadIdx.x;
  {
    int row = tid >> 1, hs = tid & 1;
    const _Float16* qp = q + (size_t)(m0 + row) * 512 + h * 64 + hs * 32;
    const float* kp = ksum + bh * 64 + hs * 32;
    float s = 0.f;
#pragma unroll
    for (int j = 0; j < 32; ++j) s += (float)qp[j] * kp[j];
    s += __shfl_xor(s, 1);
    if (hs == 0) sden[row] = s + 1e-5f;
  }
  __syncthreads();

  int wave = tid >> 5, lane = tid & 31, lr = lane & 15, ks = lane >> 4;
  int mw = m0 + wave * 16;
  v8f acc[4] = {};
#pragma unroll
  for (int k0 = 0; k0 < 64; k0 += 32) {
    const _Float16* ap = q + (size_t)(mw + lr) * 512 + h * 64 + k0 + ks * 8;
    v16h aF = mkfrag(*(const v8h*)ap, *(const v8h*)(ap + 16));
    v16h bF[4];
#pragma unroll
    for (int et = 0; et < 4; ++et) {
      const _Float16* bp = kvT + (size_t)bh * 4096 + (et * 16 + lr) * 64 + k0 + ks * 8;
      bF[et] = mkfrag(*(const v8h*)bp, *(const v8h*)(bp + 16));
    }
#pragma unroll
    for (int et = 0; et < 4; ++et)
      acc[et] = wmma_f16(aF, bF[et], acc[et]);
  }
#pragma unroll
  for (int et = 0; et < 4; ++et)
#pragma unroll
    for (int r = 0; r < 8; ++r) {
      int m = mw + ks * 8 + r;
      att[(size_t)m * 512 + h * 64 + et * 16 + lr] = (_Float16)(acc[et][r] / sden[m - m0]);
    }
}

// ----------------------------------------------------------------
extern "C" void kernel_launch(void* const* d_in, const int* in_sizes, int n_in,
                              void* d_out, int out_size, void* d_ws, size_t ws_size,
                              hipStream_t stream) {
  const float* x   = (const float*)d_in[0];
  const float* Wq  = (const float*)d_in[1];
  const float* bq  = (const float*)d_in[2];
  const float* Wk  = (const float*)d_in[3];
  const float* bk  = (const float*)d_in[4];
  const float* Wv  = (const float*)d_in[5];
  const float* bv  = (const float*)d_in[6];
  const float* Wo  = (const float*)d_in[7];
  const float* bo  = (const float*)d_in[8];
  const float* W1  = (const float*)d_in[9];
  const float* b1  = (const float*)d_in[10];
  const float* W2  = (const float*)d_in[11];
  const float* b2  = (const float*)d_in[12];
  const float* g1  = (const float*)d_in[13];
  const float* be1 = (const float*)d_in[14];
  const float* g2  = (const float*)d_in[15];
  const float* be2 = (const float*)d_in[16];
  float* out = (float*)d_out;

  char* w = (char*)d_ws;
  auto alloc = [&](size_t bytes) {
    char* p = w;
    w += (bytes + 255) & ~(size_t)255;
    return p;
  };
  _Float16* WqkvT = (_Float16*)alloc((size_t)1536 * 512 * 2);
  _Float16* WoT   = (_Float16*)alloc((size_t)512 * 512 * 2);
  _Float16* W1T   = (_Float16*)alloc((size_t)1024 * 512 * 2);
  _Float16* W2T   = (_Float16*)alloc((size_t)512 * 1024 * 2);
  _Float16* xn16  = (_Float16*)alloc((size_t)SEQB * 512 * 2);
  _Float16* q16   = (_Float16*)alloc((size_t)SEQB * 512 * 2);
  _Float16* kT16  = (_Float16*)alloc((size_t)SEQB * 512 * 2);
  _Float16* vT16  = (_Float16*)alloc((size_t)SEQB * 512 * 2);
  _Float16* kvT16 = (_Float16*)alloc((size_t)32 * 64 * 64 * 2);
  float*    ksum  = (float*)alloc((size_t)32 * 64 * 4);
  float*    kvp   = (float*)alloc((size_t)KVCH * 32 * 4096 * 4);
  float*    ksp   = (float*)alloc((size_t)KVCH * 32 * 64 * 4);
  _Float16* att16 = (_Float16*)alloc((size_t)SEQB * 512 * 2);
  _Float16* hn16  = (_Float16*)alloc((size_t)SEQB * 512 * 2);
  _Float16* h116  = (_Float16*)alloc((size_t)SEQB * 1024 * 2);

  // weights -> transposed f16
  tcast_kernel<<<512 * 512 / 256, 256, 0, stream>>>(Wq, WqkvT,                 512, 512);
  tcast_kernel<<<512 * 512 / 256, 256, 0, stream>>>(Wk, WqkvT + 512 * 512,     512, 512);
  tcast_kernel<<<512 * 512 / 256, 256, 0, stream>>>(Wv, WqkvT + 2 * 512 * 512, 512, 512);
  tcast_kernel<<<512 * 512 / 256, 256, 0, stream>>>(Wo, WoT, 512, 512);
  tcast_kernel<<<512 * 1024 / 256, 256, 0, stream>>>(W1, W1T, 512, 1024);
  tcast_kernel<<<1024 * 512 / 256, 256, 0, stream>>>(W2, W2T, 1024, 512);

  // LN1
  ln_kernel<<<SEQB / 8, 256, 0, stream>>>(x, g1, be1, xn16);

  // QKV fused GEMM + elu epilogue
  {
    GemmP p{};
    p.A = xn16; p.Bt = WqkvT; p.N = 1536; p.K = 512; p.lda = 512; p.ldbt = 512;
    p.b0 = bq; p.b1 = bk; p.b2 = bv;
    p.o0 = q16; p.o1 = kT16; p.o2 = vT16;
    gemm_kernel<0><<<dim3(1536 / BN, SEQB / BM), 256, 0, stream>>>(p);
  }

  // per-head k^T v and sum(k): K-split partials + deterministic reduce
  kv_partial_kernel<<<dim3(32, KVCH), 128, 0, stream>>>(kT16, vT16, kvp, ksp);
  kv_reduce_kernel<<<32, 256, 0, stream>>>(kvp, ksp, kvT16, ksum);

  // att = q kv / (q.ksum + eps)
  attn_kernel<<<dim3(8, SEQB / 128), 256, 0, stream>>>(q16, kvT16, ksum, att16);

  // Wo projection + bias + residual -> d_out (f32, this is res2)
  {
    GemmP p{};
    p.A = att16; p.Bt = WoT; p.N = 512; p.K = 512; p.lda = 512; p.ldbt = 512;
    p.b0 = bo; p.resid = x; p.outF = out;
    gemm_kernel<1><<<dim3(512 / BN, SEQB / BM), 256, 0, stream>>>(p);
  }

  // LN2
  ln_kernel<<<SEQB / 8, 256, 0, stream>>>(out, g2, be2, hn16);

  // FFN1: relu(hn @ W1 + b1)
  {
    GemmP p{};
    p.A = hn16; p.Bt = W1T; p.N = 1024; p.K = 512; p.lda = 512; p.ldbt = 512;
    p.b0 = b1; p.o0 = h116;
    gemm_kernel<2><<<dim3(1024 / BN, SEQB / BM), 256, 0, stream>>>(p);
  }

  // FFN2: d_out += h1 @ W2 + b2
  {
    GemmP p{};
    p.A = h116; p.Bt = W2T; p.N = 512; p.K = 1024; p.lda = 1024; p.ldbt = 1024;
    p.b0 = b2; p.outF = out;
    gemm_kernel<3><<<dim3(512 / BN, SEQB / BM), 256, 0, stream>>>(p);
  }
}
